// RelativeMultiHeadAttention_21500606284388
// MI455X (gfx1250) — compile-verified, matrix operand order checked by reading, not run on hardware
//
#include <hip/hip_runtime.h>
#include <hip/hip_bf16.h>

// ---------------------------------------------------------------------------
// Relative Multi-Head Attention (Transformer-XL style) for gfx1250 / MI455X.
// All GEMMs via V_WMMA_F32_16X16X32_BF16 (wave32, fp32 accum).
// Big GEMMs: block-level 16x512 row-panel macro-tiles, A panel staged in LDS
// via the Tensor Data Mover (tensor_load_to_lds + s_wait_tensorcnt), converted
// fp32->bf16 exactly once, 4 WMMA accumulators per wave.
// ---------------------------------------------------------------------------

typedef __attribute__((ext_vector_type(16))) __bf16 v16bf;
typedef __attribute__((ext_vector_type(8)))  float  v8f;
typedef unsigned int u32x4 __attribute__((ext_vector_type(4)));
typedef int          i32x8 __attribute__((ext_vector_type(8)));
typedef int          i32x4 __attribute__((ext_vector_type(4)));

#define DEV __device__ __forceinline__

constexpr int NB  = 8;    // batch
constexpr int NH  = 8;    // heads
constexpr int SEQ = 512;  // q_len == k_len
constexpr int DM  = 512;  // d_model
constexpr int DK  = 64;   // head dim

#ifndef USE_TDM
#define USE_TDM 1
#endif

// ---- bf16 packing (round-to-nearest-even) ---------------------------------
DEV unsigned pk_bf16(float a, float b) {
  unsigned ua = __float_as_uint(a), ub = __float_as_uint(b);
  unsigned ra = (ua + 0x7FFFu + ((ua >> 16) & 1u)) >> 16;
  unsigned rb = (ub + 0x7FFFu + ((ub >> 16) & 1u)) >> 16;
  return (ra & 0xFFFFu) | (rb << 16);
}
DEV unsigned short bf16_1(float f) {
  unsigned x = __float_as_uint(f);
  return (unsigned short)((x + 0x7FFFu + ((x >> 16) & 1u)) >> 16);
}

// ---- WMMA fragment loaders -------------------------------------------------
// A/B 16-bit fragment layout (ISA 7.12.2): lane L handles row (L&15); VGPR j
// holds the bf16 pair at K = (j<4 ? 2j : 16+2(j-4)) + 8*(L>>4).
DEV int k0_of(int j) { return ((j & 3) * 2) + ((j >> 2) * 16); }

DEV v16bf frag_f32(const float* __restrict__ p, long row, long ld, int kk, int lane) {
  int half = (lane >> 4) & 1;
  union { unsigned u[8]; v16bf v; } r;
  const float* base = p + row * ld + kk + half * 8;
#pragma unroll
  for (int j = 0; j < 8; ++j) {
    float2 f = *(const float2*)(base + k0_of(j));
    r.u[j] = pk_bf16(f.x, f.y);
  }
  return r.v;
}

DEV v16bf frag_bf16(const unsigned short* __restrict__ p, long row, long ld, int kk, int lane) {
  int half = (lane >> 4) & 1;
  union { unsigned u[8]; v16bf v; } r;
  const unsigned short* base = p + row * ld + kk + half * 8;
#pragma unroll
  for (int j = 0; j < 8; ++j)
    r.u[j] = *(const unsigned*)(base + k0_of(j));
  return r.v;
}

DEV v8f wmma_bf16(v16bf a, v16bf b, v8f c) {
  return __builtin_amdgcn_wmma_f32_16x16x32_bf16(false, a, false, b, (short)0, c,
                                                 false, false);
}

// ---------------------------------------------------------------------------
// Kernel 0: fp32 -> bf16 conversion (for the five 512x512 weight matrices).
// ---------------------------------------------------------------------------
__global__ void k_cvt(const float* __restrict__ in, unsigned* __restrict__ out, long n4) {
  long i = (long)blockIdx.x * blockDim.x + threadIdx.x;
  if (i < n4) {
    float4 f = ((const float4*)in)[i];
    out[i * 2 + 0] = pk_bf16(f.x, f.y);
    out[i * 2 + 1] = pk_bf16(f.z, f.w);
  }
}

// ---------------------------------------------------------------------------
// Kernel 1: macro-tiled  C = X(f32, Mx512) @ Wb(bf16, 512x512)^T.
// One block = one 16x512 output row-panel. A panel (16x512 f32, 32KB) staged
// in LDS (TDM when available), converted to bf16 once, shared by 8 waves.
// Each wave: 4 n-tiles (16x64), A fragment reused 4x.
// layout 0: out[((b*NH+h)*SEQ+s)*DK+d]   (m=b*SEQ+s, n=h*DK+d)
// layout 1: out[((b*NH+h)*DK+d)*SEQ+s]
// layout 2: out[m*DM+n]
// ---------------------------------------------------------------------------
struct SMemPanel {
  float    a_f32[16 * 512];   // 32 KB, at LDS offset 0 (TDM destination)
  unsigned a_bf[16 * 512 / 2];  // 16 KB packed bf16
};

__global__ void k_gemm_macro(const float* __restrict__ X, const unsigned short* __restrict__ Wb,
                             long M,
                             unsigned short* __restrict__ out0, const float* __restrict__ add0,
                             unsigned short* __restrict__ out1, const float* __restrict__ add1,
                             int layout) {
  __shared__ SMemPanel sm;
  const int tid  = threadIdx.x;
  const int lane = tid & 31;
  const int wib  = tid >> 5;            // wave-in-block, 0..7
  const long tm  = blockIdx.x;          // row-panel index
  const float* src = X + tm * (16 * 512);

  // ---- stage A panel (16x512 fp32 = 8192 floats) into LDS ----
#if USE_TDM && __has_builtin(__builtin_amdgcn_tensor_load_to_lds) && \
    __has_builtin(__builtin_amdgcn_s_wait_tensorcnt)
  if (wib == 0) {
    unsigned long long ga = (unsigned long long)src;
    // D# group0: count=1, lds_addr=0 (sm.a_f32 is first in LDS), type=2.
    u32x4 g0 = { 1u, 0u, (unsigned)(ga & 0xFFFFFFFFu),
                 (unsigned)(((ga >> 32) & 0x1FFFFFFu) | (2u << 30)) };
    // D# group1: data_size=4B (code 2); 2-D tile 512x16 of a 512xM tensor.
    unsigned td0 = 512u, td1 = (unsigned)M, t0 = 512u, t1 = 16u, s0 = 512u;
    i32x8 g1 = { (int)(2u << 16),
                 (int)((td0 & 0xFFFFu) << 16),
                 (int)((td0 >> 16) | ((td1 & 0xFFFFu) << 16)),
                 (int)(((td1 >> 16) & 0xFFFFu) | (t0 << 16)),
                 (int)(t1 & 0xFFFFu),
                 (int)s0, 0, 0 };
    i32x4 z4 = { 0, 0, 0, 0 };
#if __clang_major__ >= 23
    i32x8 z8 = { 0, 0, 0, 0, 0, 0, 0, 0 };
    __builtin_amdgcn_tensor_load_to_lds(g0, g1, z4, z4, z8, 0);
#else
    __builtin_amdgcn_tensor_load_to_lds(g0, g1, z4, z4, 0);
#endif
    __builtin_amdgcn_s_wait_tensorcnt(0);
  }
#else
#pragma unroll
  for (int i = 0; i < 8; ++i)  // 256 thr x 8 float4 = 8192 floats
    ((float4*)sm.a_f32)[tid + i * 256] = ((const float4*)src)[tid + i * 256];
#endif
  __syncthreads();

  // ---- convert fp32 panel -> packed bf16 panel (once per element) ----
#pragma unroll
  for (int i = 0; i < 16; ++i) {        // 4096 dword-pairs
    int p = tid + i * 256;
    float2 f = ((const float2*)sm.a_f32)[p];
    sm.a_bf[p] = pk_bf16(f.x, f.y);
  }
  __syncthreads();

  // ---- compute: wave wib covers n in [wib*64, wib*64+64) ----
  const unsigned short* ldsA = (const unsigned short*)sm.a_bf;
  v8f acc[4] = {};
  for (int kk = 0; kk < 512; kk += 32) {
    v16bf a = frag_bf16(ldsA, lane & 15, 512, kk, lane);
#pragma unroll
    for (int j = 0; j < 4; ++j) {
      long rowB = (wib * 4 + j) * 16 + (lane & 15);
      v16bf b = frag_bf16(Wb, rowB, 512, kk, lane);
      acc[j] = wmma_bf16(a, b, acc[j]);
    }
  }

  // ---- store ----
  int half = lane >> 4;
#pragma unroll
  for (int j = 0; j < 4; ++j) {
    int n = (wib * 4 + j) * 16 + (lane & 15);
#pragma unroll
    for (int r = 0; r < 8; ++r) {
      long m = tm * 16 + r + 8 * half;
      long idx;
      if (layout == 2) {
        idx = m * DM + n;
      } else {
        long bb = m / SEQ, s = m % SEQ;
        int  h = n / DK, d = n % DK;
        idx = (layout == 0) ? (((bb * NH + h) * SEQ + s) * (long)DK + d)
                            : (((bb * NH + h) * DK + d) * (long)SEQ + s);
      }
      float v0 = acc[j][r] + (add0 ? add0[n] : 0.f);
      out0[idx] = bf16_1(v0);
      if (out1) out1[idx] = bf16_1(acc[j][r] + add1[n]);
    }
  }
}

// ---------------------------------------------------------------------------
// Kernel 2: content scores  S[b,h,q,k] = Qc[b,h,q,:] . Kh[b,h,k,:]
// ---------------------------------------------------------------------------
__global__ void k_content(const unsigned short* __restrict__ Qc,
                          const unsigned short* __restrict__ Kh,
                          float* __restrict__ scores) {
  int lane = threadIdx.x & 31;
  int wave = (blockIdx.x * blockDim.x + threadIdx.x) >> 5;
  int kt = wave & 31, qt = (wave >> 5) & 31, h = (wave >> 10) & 7, b = (wave >> 13) & 7;

  long base = (long)(b * NH + h) * SEQ * DK;
  long rowA = qt * 16 + (lane & 15);
  long rowB = kt * 16 + (lane & 15);
  v8f acc = {};
#pragma unroll
  for (int kk = 0; kk < DK; kk += 32) {
    v16bf a  = frag_bf16(Qc + base, rowA, DK, kk, lane);
    v16bf bb = frag_bf16(Kh + base, rowB, DK, kk, lane);
    acc = wmma_bf16(a, bb, acc);
  }
  long sb = (long)(b * NH + h) * SEQ * SEQ;
  int half = lane >> 4, n = kt * 16 + (lane & 15);
#pragma unroll
  for (int r = 0; r < 8; ++r) {
    long m = qt * 16 + r + 8 * half;
    scores[sb + m * SEQ + n] = acc[r];
  }
}

// ---------------------------------------------------------------------------
// Kernel 3: position scores.  For fixed (h,q): P = Qp[:,h,q,:](8x64, pad 16)
//           @ Rh[h,q,:,:]^T(64x512) — batch is the M dim, so the 256 MB R is
//           streamed exactly once. Accumulates into scores.
// ---------------------------------------------------------------------------
__global__ void k_position(const unsigned short* __restrict__ Qp,
                           const unsigned short* __restrict__ Rb,
                           float* __restrict__ scores) {
  int lane = threadIdx.x & 31;
  int wave = (blockIdx.x * blockDim.x + threadIdx.x) >> 5;
  int kt = wave & 31, q = (wave >> 5) & 511, h = (wave >> 14) & 7;

  int m = lane & 15, half = lane >> 4;
  int bm = (m < NB) ? m : 0;                   // clamp; zeroed after load
  long rowB = kt * 16 + (lane & 15);           // k index
  const unsigned short* bptr = Rb + ((long)rowB * SEQ + q) * DM + h * DK;

  v8f acc = {};
#pragma unroll
  for (int kk = 0; kk < DK; kk += 32) {
    union { unsigned u[8]; v16bf v; } A;
    const unsigned short* qb =
        Qp + (((long)bm * NH + h) * SEQ + q) * DK + kk + half * 8;
#pragma unroll
    for (int j = 0; j < 8; ++j) {
      unsigned val = *(const unsigned*)(qb + k0_of(j));
      A.u[j] = (m < NB) ? val : 0u;            // padded M rows are zero
    }
    v16bf bb = frag_bf16(bptr, 0, 0, kk, lane);
    acc = wmma_bf16(A.v, bb, acc);
  }

  int n = kt * 16 + (lane & 15);
  if (half == 0) {                             // rows m = r <=> b = 0..7
#pragma unroll
    for (int r = 0; r < 8; ++r) {
      long idx = ((long)(r * NH + h) * SEQ + q) * SEQ + n;
      scores[idx] += acc[r];
    }
  }
}

// ---------------------------------------------------------------------------
// Kernel 4: scale (1/8), causal mask (k>q -> -1e9), softmax. One wave per row.
// ---------------------------------------------------------------------------
__global__ void k_softmax(float* __restrict__ scores) {
  int lane = threadIdx.x & 31;
  int wid = (blockIdx.x * blockDim.x + threadIdx.x) >> 5;
  int q = wid & 511;
  float* row = scores + (long)wid * SEQ;

  float vals[16];
  float mx = -3.4e38f;
#pragma unroll
  for (int i = 0; i < 16; ++i) {
    int k = lane + i * 32;
    float s = row[k] * 0.125f;
    if (k > q) s = -1e9f;
    vals[i] = s;
    mx = fmaxf(mx, s);
  }
#pragma unroll
  for (int off = 16; off; off >>= 1) mx = fmaxf(mx, __shfl_xor(mx, off, 32));
  float sum = 0.f;
#pragma unroll
  for (int i = 0; i < 16; ++i) { vals[i] = __expf(vals[i] - mx); sum += vals[i]; }
#pragma unroll
  for (int off = 16; off; off >>= 1) sum += __shfl_xor(sum, off, 32);
  float inv = 1.f / sum;
#pragma unroll
  for (int i = 0; i < 16; ++i) row[lane + i * 32] = vals[i] * inv;
}

// ---------------------------------------------------------------------------
// Kernel 5: O[b,h,q,:] = weights[b,h,q,:] @ Vh. Wave computes 16q x 64d:
// A fragment (weights fp32->bf16) converted once, reused by 4 accumulators.
// Vh is stored transposed (B,H,DK,S) so B-fragment reads are row-contiguous.
// ---------------------------------------------------------------------------
__global__ void k_av(const float* __restrict__ Wt, const unsigned short* __restrict__ VhT,
                     unsigned short* __restrict__ AO) {
  int lane = threadIdx.x & 31;
  int wave = (blockIdx.x * blockDim.x + threadIdx.x) >> 5;
  int qt = wave & 31, h = (wave >> 5) & 7, b = (wave >> 8) & 7;

  long wbase = (long)(b * NH + h) * SEQ * SEQ;
  long vbase = (long)(b * NH + h) * DK * SEQ;
  long rowA = qt * 16 + (lane & 15);
  v8f acc[4] = {};
  for (int kk = 0; kk < SEQ; kk += 32) {
    if (kk + 32 < SEQ)
      __builtin_prefetch(Wt + wbase + rowA * SEQ + kk + 32, 0, 3);
    v16bf a = frag_f32(Wt + wbase, rowA, SEQ, kk, lane);
#pragma unroll
    for (int j = 0; j < 4; ++j) {
      long rowB = j * 16 + (lane & 15);
      v16bf bb = frag_bf16(VhT + vbase, rowB, SEQ, kk, lane);
      acc[j] = wmma_bf16(a, bb, acc[j]);
    }
  }
  int half = lane >> 4;
#pragma unroll
  for (int j = 0; j < 4; ++j) {
    int d = j * 16 + (lane & 15);
#pragma unroll
    for (int r = 0; r < 8; ++r) {
      long qrow = qt * 16 + r + 8 * half;
      AO[((long)b * SEQ + qrow) * DM + h * DK + d] = bf16_1(acc[j][r]);
    }
  }
}

// ---------------------------------------------------------------------------
// Kernel 6: out = AO(4096x512 bf16) @ Wo_b(bf16)^T -> fp32. Wave: 16x64.
// ---------------------------------------------------------------------------
__global__ void k_final(const unsigned short* __restrict__ AO,
                        const unsigned short* __restrict__ Wob,
                        float* __restrict__ out) {
  int lane = threadIdx.x & 31;
  int wave = (blockIdx.x * blockDim.x + threadIdx.x) >> 5;
  int nb = wave & 7, tm = wave >> 3;

  long rowA = (long)tm * 16 + (lane & 15);
  v8f acc[4] = {};
  for (int kk = 0; kk < DM; kk += 32) {
    v16bf a = frag_bf16(AO, rowA, DM, kk, lane);
#pragma unroll
    for (int j = 0; j < 4; ++j) {
      long rowB = (nb * 4 + j) * 16 + (lane & 15);
      v16bf b = frag_bf16(Wob, rowB, DM, kk, lane);
      acc[j] = wmma_bf16(a, b, acc[j]);
    }
  }
  int half = lane >> 4;
#pragma unroll
  for (int j = 0; j < 4; ++j) {
    int n = (nb * 4 + j) * 16 + (lane & 15);
#pragma unroll
    for (int r = 0; r < 8; ++r) {
      long m = (long)tm * 16 + r + 8 * half;
      out[m * DM + n] = acc[j][r];
    }
  }
}

// ---------------------------------------------------------------------------
extern "C" void kernel_launch(void* const* d_in, const int* in_sizes, int n_in,
                              void* d_out, int out_size, void* d_ws, size_t ws_size,
                              hipStream_t stream) {
  const float* query = (const float*)d_in[0];
  const float* key_  = (const float*)d_in[1];
  const float* value = (const float*)d_in[2];
  const float* rel   = (const float*)d_in[3];
  // d_in[4] = mask (tril) — applied analytically in k_softmax
  const float* Wq = (const float*)d_in[5];
  const float* Wk = (const float*)d_in[6];
  const float* Wv = (const float*)d_in[7];
  const float* Wo = (const float*)d_in[8];
  const float* Wr = (const float*)d_in[9];
  const float* u  = (const float*)d_in[10];
  const float* v  = (const float*)d_in[11];

  float* out     = (float*)d_out;
  float* weights = out + (long)NB * SEQ * DM;   // second tuple element

  // Workspace (bf16): 5x4MB staging + 2.5MB weights + 256MB projected R.
  unsigned short* ws  = (unsigned short*)d_ws;
  const long NE = (long)NB * NH * SEQ * DK;     // 2,097,152
  const long NW = (long)DM * DM;                // 262,144
  unsigned short* Qc  = ws;             // (B,H,S,DK)  Qh + u
  unsigned short* Qp  = Qc + NE;        // (B,H,S,DK)  Qh + v
  unsigned short* Kh  = Qp + NE;        // (B,H,S,DK)
  unsigned short* VhT = Kh + NE;        // (B,H,DK,S)
  unsigned short* AO  = VhT + NE;       // (B,S,DM)
  unsigned short* Wqb = AO + NE;
  unsigned short* Wkb = Wqb + NW;
  unsigned short* Wvb = Wkb + NW;
  unsigned short* Wob = Wvb + NW;
  unsigned short* Wrb = Wob + NW;
  unsigned short* Rb  = Wrb + NW;       // (K,Q,DM) bf16, 256 MB

  dim3 blk(256);

  // Weight conversions (once): 65536 float4 each -> 256 blocks.
  k_cvt<<<256, blk, 0, stream>>>(Wq, (unsigned*)Wqb, NW / 4);
  k_cvt<<<256, blk, 0, stream>>>(Wk, (unsigned*)Wkb, NW / 4);
  k_cvt<<<256, blk, 0, stream>>>(Wv, (unsigned*)Wvb, NW / 4);
  k_cvt<<<256, blk, 0, stream>>>(Wo, (unsigned*)Wob, NW / 4);
  k_cvt<<<256, blk, 0, stream>>>(Wr, (unsigned*)Wrb, NW / 4);

  // Projections: one block per 16-row panel.
  k_gemm_macro<<<256,   blk, 0, stream>>>(query, Wqb, 4096, Qc, u, Qp, v, 0);
  k_gemm_macro<<<256,   blk, 0, stream>>>(key_,  Wkb, 4096, Kh, nullptr, nullptr, nullptr, 0);
  k_gemm_macro<<<256,   blk, 0, stream>>>(value, Wvb, 4096, VhT, nullptr, nullptr, nullptr, 1);
  // R projection: 262144 rows -> 16384 blocks (137 GFLOP, dominant term).
  k_gemm_macro<<<16384, blk, 0, stream>>>(rel, Wrb, (long)SEQ * SEQ, Rb,
                                          nullptr, nullptr, nullptr, 2);

  k_content <<<8192,  blk, 0, stream>>>(Qc, Kh, weights);
  k_position<<<16384, blk, 0, stream>>>(Qp, Rb, weights);
  k_softmax <<<4096,  blk, 0, stream>>>(weights);
  k_av      <<<256,   blk, 0, stream>>>(weights, VhT, AO);
  k_final   <<<256,   blk, 0, stream>>>(AO, Wob, out);
}
